// HQQDORA_4243427688520
// MI455X (gfx1250) — matrix operand order repaired and checked by reading, not compiled, not run on hardware
//
#include <hip/hip_runtime.h>
#include <hip/hip_bf16.h>

// ---------------------------------------------------------------------------
// HQQ-DoRA: out[m,n] = (x·Wᵀ + (x·Aᵀ)·Bᵀ)[m,n] * magnitude[n] / ‖W+B·A‖₂[n]
// M=8192, N=4096, K=4096, R=32, fp32 in/out. Matrix-core bound (275 GFLOP)
// -> bf16 WMMA path with one-time f32->bf16 precopy, async memory->LDS
// staging, double-buffered LDS, lora folded in as a final K=32 step.
// ---------------------------------------------------------------------------
#define GM 8192
#define GN 4096
#define GK 4096
#define GR 32

typedef __attribute__((ext_vector_type(16))) __bf16 v16bf;
typedef __attribute__((ext_vector_type(8)))  __bf16 v8bf;
typedef __attribute__((ext_vector_type(8)))  float  v8f;

// Round-to-nearest-even f32 -> bf16 (used only in the one-shot precopy kernels)
__device__ __forceinline__ __bf16 f2bf(float f) {
    union { float f; unsigned u; } v; v.f = f;
    unsigned r = v.u + 0x7FFFu + ((v.u >> 16) & 1u);
    unsigned short h = (unsigned short)(r >> 16);
    __bf16 b;
    __builtin_memcpy(&b, &h, 2);
    return b;
}

__device__ __forceinline__ v8bf pack8(float4 a, float4 b) {
    v8bf r;
    r[0] = f2bf(a.x); r[1] = f2bf(a.y); r[2] = f2bf(a.z); r[3] = f2bf(a.w);
    r[4] = f2bf(b.x); r[5] = f2bf(b.y); r[6] = f2bf(b.z); r[7] = f2bf(b.w);
    return r;
}

// LDS byte-offset of a generic pointer that points into LDS
__device__ __forceinline__ unsigned lds_off_of(const void* p) {
    return (unsigned)(unsigned long long)
           (__attribute__((address_space(3))) const void*)p;
}

// 64B memory->LDS async copy (four b128 beats; offset applies to both sides)
__device__ __forceinline__ void async_cp64(const __bf16* g, __bf16* l) {
    unsigned loff = lds_off_of(l);
    asm volatile(
        "global_load_async_to_lds_b128 %0, %1, off\n\t"
        "global_load_async_to_lds_b128 %0, %1, off offset:16\n\t"
        "global_load_async_to_lds_b128 %0, %1, off offset:32\n\t"
        "global_load_async_to_lds_b128 %0, %1, off offset:48"
        :: "v"(loff), "v"(g) : "memory");
}
// 32B memory->LDS async copy
__device__ __forceinline__ void async_cp32(const __bf16* g, __bf16* l) {
    unsigned loff = lds_off_of(l);
    asm volatile(
        "global_load_async_to_lds_b128 %0, %1, off\n\t"
        "global_load_async_to_lds_b128 %0, %1, off offset:16"
        :: "v"(loff), "v"(g) : "memory");
}
__device__ __forceinline__ void async_wait0() {
    asm volatile("s_wait_asynccnt 0x0" ::: "memory");
}

// ---------------------------------------------------------------------------
// f32 -> bf16 bulk convert (8 elems/thread)
// ---------------------------------------------------------------------------
__global__ __launch_bounds__(256) void cvt_bf16_kernel(
    const float* __restrict__ src, __bf16* __restrict__ dst) {
    const size_t i = ((size_t)blockIdx.x * 256 + threadIdx.x) * 8;
    const float4 a = ((const float4*)(src + i))[0];
    const float4 b = ((const float4*)(src + i))[1];
    *(v8bf*)(dst + i) = pack8(a, b);
}

// ---------------------------------------------------------------------------
// scale[o] = magnitude[o] / || W[o,:] + (loraB @ loraA)[o,:] ||_2
// ---------------------------------------------------------------------------
__global__ __launch_bounds__(256) void dora_scale_kernel(
    const float* __restrict__ W, const float* __restrict__ A,
    const float* __restrict__ Bm, const float* __restrict__ mag,
    float* __restrict__ scale) {
    __shared__ float sB[GR];
    __shared__ float red[256];
    const int o = blockIdx.x;
    const int tid = threadIdx.x;
    if (tid < GR) sB[tid] = Bm[(size_t)o * GR + tid];
    __syncthreads();

    float acc = 0.f;
    for (int i = tid; i < GK; i += 256) {
        float ba = 0.f;
        #pragma unroll
        for (int r = 0; r < GR; ++r) ba += sB[r] * A[(size_t)r * GK + i];
        const float v = W[(size_t)o * GK + i] + ba;
        acc += v * v;
    }
    red[tid] = acc;
    __syncthreads();
    for (int s = 128; s > 0; s >>= 1) {
        if (tid < s) red[tid] += red[tid + s];
        __syncthreads();
    }
    if (tid == 0) scale[o] = mag[o] / sqrtf(red[0]);
}

// ---------------------------------------------------------------------------
// xa_bf[s,r] = bf16( dot(x[s,:], loraA[r,:]) )
// ---------------------------------------------------------------------------
__global__ __launch_bounds__(256) void dora_xa_kernel(
    const float* __restrict__ x, const float* __restrict__ A,
    __bf16* __restrict__ xabf) {
    __shared__ float xs[GK];
    __shared__ float part[8][GR];
    const int s = blockIdx.x;
    const int tid = threadIdx.x;
    const float* xrow = x + (size_t)s * GK;
    for (int i = tid; i < GK; i += 256) xs[i] = xrow[i];
    __syncthreads();

    const int r = tid & 31;
    const int g = tid >> 5;
    const float* arow = A + (size_t)r * GK;
    float p = 0.f;
    const int i0 = g * (GK / 8);
    for (int i = i0; i < i0 + (GK / 8); ++i) p += xs[i] * arow[i];
    part[g][r] = p;
    __syncthreads();
    if (tid < GR) {
        float t = 0.f;
        #pragma unroll
        for (int gg = 0; gg < 8; ++gg) t += part[gg][tid];
        xabf[(size_t)s * GR + tid] = f2bf(t);
    }
}

// ---------------------------------------------------------------------------
// Main fused GEMM (bf16 inputs from workspace).
//   WG tile 128x128, 8 waves (4M x 2N), wave tile 32x64 = 2x4 WMMA tiles.
//   K-chunk 64 = two WMMA sub-steps -> 16 WMMAs per barrier.
//   Double-buffered LDS (72KB), K-loop unrolled x2 so buffer index is a
//   compile-time constant. B fragments loaded one-at-a-time to keep peak
//   VGPR pressure low (accum 64 + A 16 + B 8) and avoid accumulator copies.
//   Lora (K=R=32) peeled as a final single-substep iteration.
// ---------------------------------------------------------------------------
#define TILE_M 128
#define TILE_N 128
#define KTILE  64
#define LDT    72          // padded bf16 row stride (144B rows, 16B aligned,
                           // 36-dword stride -> conflict-free fragment loads)
#define KB_MAIN (GK / KTILE)   // 64 main K-chunks (even)

__global__ __launch_bounds__(256) void dora_gemm_kernel(
    const __bf16* __restrict__ xbf,  const __bf16* __restrict__ wbf,
    const __bf16* __restrict__ xabf, const __bf16* __restrict__ lbbf,
    const float* __restrict__ scale, float* __restrict__ out) {
    __shared__ __align__(16) __bf16 As[2][TILE_M * LDT];
    __shared__ __align__(16) __bf16 Bs[2][TILE_N * LDT];

    const int tid   = threadIdx.x;
    const int lane  = tid & 31;
    const int wave  = tid >> 5;
    const int waveM = wave >> 1;     // 0..3  (rows, 32 each)
    const int waveN = wave & 1;      // 0..1  (cols, 64 each)
    const int hl    = lane >> 4;
    const int lrow  = lane & 15;

    const int m0 = blockIdx.y * TILE_M;
    const int n0 = blockIdx.x * TILE_N;

    // staging: 128 rows x 2 threads/row, 32 bf16 (64B) per thread per tile
    const int srow  = tid >> 1;
    const int shalf = tid & 1;

    auto stage_main = [&](int kb, int buf) {
        const __bf16* ga = xbf + (size_t)(m0 + srow) * GK + kb * KTILE + shalf * 32;
        const __bf16* gb = wbf + (size_t)(n0 + srow) * GK + kb * KTILE + shalf * 32;
        async_cp64(ga, &As[buf][srow * LDT + shalf * 32]);
        async_cp64(gb, &Bs[buf][srow * LDT + shalf * 32]);
    };
    auto stage_lora = [&](int buf) {   // K=R=32: only low half of each row
        if (shalf == 0) {
            async_cp32(xabf + (size_t)(m0 + srow) * GR, &As[buf][srow * LDT]);
            async_cp32(lbbf + (size_t)(n0 + srow) * GR, &Bs[buf][srow * LDT]);
        }
    };

    v8f c[2][4];
    #pragma unroll
    for (int i = 0; i < 2; ++i)
        #pragma unroll
        for (int j = 0; j < 4; ++j)
            c[i][j] = (v8f){0.f, 0.f, 0.f, 0.f, 0.f, 0.f, 0.f, 0.f};

    // one WMMA sub-step (K=32 slice at column offset ks*32 inside the chunk).
    // B fragments are loaded one at a time and consumed immediately to keep
    // the live register set small.
    auto compute = [&](int buf, int ks) {
        // A 16x32 bf16 frag: lane (hl, m=lrow): e0..7 = K[8hl..8hl+7],
        //                                       e8..15 = K[16+8hl..23+8hl]
        v16bf af[2];
        #pragma unroll
        for (int tm = 0; tm < 2; ++tm) {
            const __bf16* p =
                &As[buf][(waveM * 32 + tm * 16 + lrow) * LDT + ks * 32];
            v8bf lo = *(const v8bf*)(p + 8 * hl);
            v8bf hi = *(const v8bf*)(p + 16 + 8 * hl);
            af[tm] = __builtin_shufflevector(lo, hi,
                0, 1, 2, 3, 4, 5, 6, 7, 8, 9, 10, 11, 12, 13, 14, 15);
        }
        #pragma unroll
        for (int tn = 0; tn < 4; ++tn) {
            // B 32x16 bf16 frag: lane (hl, n=lrow): e0..7 = K[8hl..8hl+7],
            //                                       e8..15 = K[16+8hl..23+8hl]
            const __bf16* p =
                &Bs[buf][(waveN * 64 + tn * 16 + lrow) * LDT + ks * 32 + 8 * hl];
            v8bf lo = *(const v8bf*)p;
            v8bf hi = *(const v8bf*)(p + 16);
            v16bf bfr = __builtin_shufflevector(lo, hi,
                0, 1, 2, 3, 4, 5, 6, 7, 8, 9, 10, 11, 12, 13, 14, 15);
            c[0][tn] = __builtin_amdgcn_wmma_f32_16x16x32_bf16(
                false, af[0], false, bfr, (short)0, c[0][tn], false, false);
            c[1][tn] = __builtin_amdgcn_wmma_f32_16x16x32_bf16(
                false, af[1], false, bfr, (short)0, c[1][tn], false, false);
        }
    };

    stage_main(0, 0);

    // K-loop unrolled x2: buffer index is a compile-time constant per half.
    for (int kb2 = 0; kb2 < KB_MAIN; kb2 += 2) {
        async_wait0();
        __syncthreads();
        stage_main(kb2 + 1, 1);
        compute(0, 0);
        compute(0, 1);

        async_wait0();
        __syncthreads();
        if (kb2 + 2 < KB_MAIN) stage_main(kb2 + 2, 0);
        else                   stage_lora(0);
        compute(1, 0);
        compute(1, 1);
    }

    // final peeled iteration: low-rank adapter contribution (single sub-step)
    async_wait0();
    __syncthreads();
    compute(0, 0);

    // epilogue: *= scale[n]; C/D layout: VGPR v, half hl -> row v + 8*hl
    #pragma unroll
    for (int tn = 0; tn < 4; ++tn) {
        const int n  = n0 + waveN * 64 + tn * 16 + lrow;
        const float sc = scale[n];
        #pragma unroll
        for (int tm = 0; tm < 2; ++tm) {
            const int mbase = m0 + waveM * 32 + tm * 16 + 8 * hl;
            #pragma unroll
            for (int v = 0; v < 8; ++v)
                out[(size_t)(mbase + v) * GN + n] = c[tm][tn][v] * sc;
        }
    }
}

// ---------------------------------------------------------------------------
// Launcher.  Workspace layout:
//   scale : GN f32 (16 KB) | xa_bf : GM*GR bf16 (512 KB)
//   lb_bf : GN*GR bf16 (256 KB) | x_bf : GM*GK bf16 (64 MB)
//   w_bf  : GN*GK bf16 (32 MB)                      total ~96.8 MB
// ---------------------------------------------------------------------------
extern "C" void kernel_launch(void* const* d_in, const int* in_sizes, int n_in,
                              void* d_out, int out_size, void* d_ws, size_t ws_size,
                              hipStream_t stream) {
    const float* x     = (const float*)d_in[0];
    const float* W     = (const float*)d_in[1];
    const float* loraA = (const float*)d_in[2];
    const float* loraB = (const float*)d_in[3];
    const float* mag   = (const float*)d_in[4];
    float* out = (float*)d_out;

    float*  scale = (float*)d_ws;
    __bf16* xabf  = (__bf16*)(scale + GN);
    __bf16* lbbf  = xabf + (size_t)GM * GR;
    __bf16* xbf   = lbbf + (size_t)GN * GR;
    __bf16* wbf   = xbf  + (size_t)GM * GK;

    // one-shot precision conversion (bandwidth-bound)
    cvt_bf16_kernel<<<(size_t)GM * GK / 2048, 256, 0, stream>>>(x, xbf);
    cvt_bf16_kernel<<<(size_t)GN * GK / 2048, 256, 0, stream>>>(W, wbf);
    cvt_bf16_kernel<<<(size_t)GN * GR / 2048, 256, 0, stream>>>(loraB, lbbf);

    dora_scale_kernel<<<GN, 256, 0, stream>>>(W, loraA, loraB, mag, scale);
    dora_xa_kernel<<<GM, 256, 0, stream>>>(x, loraA, xabf);

    dim3 grid(GN / TILE_N, GM / TILE_M);   // 32 x 64
    dora_gemm_kernel<<<grid, 256, 0, stream>>>(xbf, wbf, xabf, lbbf, scale, out);
}
